// Model_DIN_V2_Gru_Vec_attGru_Neg_56186762167104
// MI455X (gfx1250) — matrix operand order repaired and checked by reading, not compiled, not run on hardware
//
#include <hip/hip_runtime.h>
#include <math.h>

typedef __attribute__((ext_vector_type(16))) _Float16 v16h;
typedef __attribute__((ext_vector_type(8)))  _Float16 v8h;
typedef __attribute__((ext_vector_type(4)))  _Float16 v4h;
typedef __attribute__((ext_vector_type(8)))  float    v8f;
typedef __attribute__((ext_vector_type(4)))  float    v4f;
typedef __attribute__((ext_vector_type(4)))  unsigned int u32x4;
typedef __attribute__((ext_vector_type(8)))  int      i32x8;
typedef __attribute__((ext_vector_type(4)))  int      i32x4;

#define WMMA_F16(a, b, c) \
    __builtin_amdgcn_wmma_f32_16x16x32_f16(false, (a), false, (b), (short)0, (c), false, false)

#if __has_builtin(__builtin_amdgcn_tensor_load_to_lds) && \
    __has_builtin(__builtin_amdgcn_s_wait_tensorcnt)
#define HAVE_TDM 1
#else
#define HAVE_TDM 0
#endif

// ---------------------------------------------------------------------------
// Fragment loaders (layouts per CDNA5 ISA §7.12.2, wave32)
// ---------------------------------------------------------------------------

// A-matrix 16x32 f16 fragment from row-major f16 LDS (stride in halves, must be
// a multiple of 8 for 16B alignment): two contiguous 8-half runs -> 2x ds_load_b128.
__device__ inline v16h load_A(const _Float16* base, int stride, int kbase) {
    const int l = threadIdx.x & 31;
    const _Float16* p = base + (l & 15) * stride + kbase + ((l >> 4) << 3);
    v8h lo = *(const v8h*)p;
    v8h hi = *(const v8h*)(p + 16);
    return __builtin_shufflevector(lo, hi, 0, 1, 2, 3, 4, 5, 6, 7,
                                           8, 9, 10, 11, 12, 13, 14, 15);
}

// B-matrix 32x16 fragment from pre-swizzled packed weights:
// packed[( (kb*NT + nt)*32 + lane )*16 + e], one aligned 32B load per lane.
__device__ inline v16h load_B_packed(const _Float16* __restrict__ P, int NT,
                                     int kb, int nt) {
    const int lane = threadIdx.x & 31;
    return *(const v16h*)(P + ((size_t)((kb * NT + nt) * 32 + lane) << 4));
}

__device__ inline float sigmoidf_(float x) { return 1.f / (1.f + __expf(-x)); }

#if HAVE_TDM
// TDM: async 2D tile load, 16 rows x 128 f32, row stride 25600 f32 (= [B,T,128]
// batch-major at fixed t), into LDS.  D# layout per CDNA5 ISA.
__device__ inline void tdm_load_x(const float* gaddr, unsigned lds_off) {
    const unsigned long long ga = (unsigned long long)(size_t)gaddr;
    u32x4 g0;
    g0[0] = 1u;                                            // count=1 (valid user D#)
    g0[1] = lds_off;                                       // lds_addr (bytes)
    g0[2] = (unsigned)(ga & 0xFFFFFFFFu);                  // global_addr[31:0]
    g0[3] = (unsigned)((ga >> 32) & 0x1FFFFFFu) | (2u << 30);  // addr[56:32] | type=2
    i32x8 g1;
    g1[0] = 0x20000;              // workgroup_mask=0, data_size=2 (4 bytes)
    g1[1] = (int)(128u << 16);    // tensor_dim0[15:0] = 128   (bits 63:48)
    g1[2] = (int)(16u << 16);     // tensor_dim1[15:0] = 16    (bits 95:80)
    g1[3] = (int)(128u << 16);    // tile_dim0 = 128           (bits 127:112)
    g1[4] = 16;                   // tile_dim1 = 16            (bits 143:128)
    g1[5] = 25600;                // tensor_dim0_stride[31:0]  (bits 191:160)
    g1[6] = 0;
    g1[7] = 0;
    i32x4 g2 = {0, 0, 0, 0};
    i32x4 g3 = {0, 0, 0, 0};
#if defined(__clang_major__) && __clang_major__ >= 23
    i32x8 gz = {};
    __builtin_amdgcn_tensor_load_to_lds(g0, g1, g2, g3, gz, 0);
#else
    __builtin_amdgcn_tensor_load_to_lds(g0, g1, g2, g3, 0);
#endif
}
#endif

// ---------------------------------------------------------------------------
// Weight pre-swizzle: f32 row-major W[K,N] -> f16 B-fragments (zero padded).
// ---------------------------------------------------------------------------
__global__ __launch_bounds__(256)
void pack_B(const float* __restrict__ W, int K, int N, int KB, int NT,
            _Float16* __restrict__ out) {
    const int total = KB * NT * 512;
    for (int i = blockIdx.x * blockDim.x + threadIdx.x; i < total;
         i += gridDim.x * blockDim.x) {
        const int e    = i & 15;
        const int lane = (i >> 4) & 31;
        const int f    = i >> 9;
        const int nt   = f % NT;
        const int kb   = f / NT;
        const int n = nt * 16 + (lane & 15);
        const int k = kb * 32 + ((lane >> 4) << 3) + (e & 7) + ((e >> 3) << 4);
        const float w = (k < K && n < N) ? W[(size_t)k * N + n] : 0.f;
        out[i] = (_Float16)w;
    }
}

// ---------------------------------------------------------------------------
// Kernel 1/4: GRU / AUGRU over time.  One WG = 16 batch rows, 8 waves.
// Wave w owns gate columns [32w,32w+32) and candidate columns [16w,16w+16).
// Weights register-resident; h state in LDS; x_t double-buffered via TDM.
// ---------------------------------------------------------------------------
template <bool AUGRU>
__global__ __launch_bounds__(256)
void gru_kernel(const float* __restrict__ X,        // [B,200,128]
                const _Float16* __restrict__ pWg,   // packed [8kb][16nt]
                const float* __restrict__ bg,
                const _Float16* __restrict__ pWc,   // packed [8kb][8nt]
                const float* __restrict__ bc,
                const int*   __restrict__ seq_len,
                const float* __restrict__ att,      // [B,200] (AUGRU only)
                float* __restrict__ out_seq,        // [B,200,128] or null
                float* __restrict__ out_final)      // [B,128] or null
{
    const int tileB = blockIdx.x * 16;
    const int tid   = threadIdx.x;
    const int wave  = tid >> 5;
    const int lane  = tid & 31;
    const int Mhi   = (lane >> 4) << 3;
    const int nloc  = lane & 15;

    __shared__ _Float16 sXH[16][256];   // [x | h]        (gates A)
    __shared__ _Float16 sXR[16][256];   // [x | rg*h]     (candidate A)
    __shared__ float    sH[16][128];    // h (f32 master)
    __shared__ float    sU[16][128];    // update gate
    __shared__ float    sC[16][128];    // candidate
#if HAVE_TDM
    __shared__ float    sXf[2][16][128];  // TDM landing buffers (f32)
#endif

    // ---- weights -> registers (one 32B vector load per fragment) ----
    v16h bgf[8][2];
    v16h bcf[8];
#pragma unroll
    for (int kb = 0; kb < 8; ++kb) {
        bgf[kb][0] = load_B_packed(pWg, 16, kb, wave * 2);
        bgf[kb][1] = load_B_packed(pWg, 16, kb, wave * 2 + 1);
        bcf[kb]    = load_B_packed(pWc, 8, kb, wave);
    }
    const float biasg0 = bg[wave * 32 + nloc];
    const float biasg1 = bg[wave * 32 + 16 + nloc];
    const float biasc  = bc[wave * 16 + nloc];

    // ---- h = 0 ----
    for (int i = tid; i < 512; i += 256) {
        const int r = i >> 5, d4 = (i & 31) << 2;
        *(v4f*)&sH[r][d4] = (v4f){0.f, 0.f, 0.f, 0.f};
        *(v4h*)&sXH[r][128 + d4] = (v4h)(_Float16)0.f;
        *(v4h*)&sXR[r][128 + d4] = (v4h)(_Float16)0.f;
    }
#if HAVE_TDM
    if (wave == 0)   // prime the pipeline: DMA x_0 into buffer 0
        tdm_load_x(&X[(size_t)tileB * 200 * 128], (unsigned)(size_t)&sXf[0][0][0]);
#endif
    __syncthreads();

    for (int t = 0; t < 200; ++t) {
#if HAVE_TDM
        if (wave == 0) __builtin_amdgcn_s_wait_tensorcnt(0);
        __syncthreads();                     // all waves: buffer t&1 is ready
        if (wave == 0 && t + 1 < 200)        // DMA next step into other buffer
            tdm_load_x(&X[((size_t)tileB * 200 + (t + 1)) * 128],
                       (unsigned)(size_t)&sXf[(t + 1) & 1][0][0]);
        const float* xsrc = &sXf[t & 1][0][0];
        for (int i = tid; i < 512; i += 256) {
            const int r = i >> 5, d4 = (i & 31) << 2;
            v4f x = *(const v4f*)(xsrc + r * 128 + d4);
            v4h xh = __builtin_convertvector(x, v4h);
            *(v4h*)&sXH[r][d4] = xh;
            *(v4h*)&sXR[r][d4] = xh;
        }
#else
        for (int i = tid; i < 512; i += 256) {
            const int r = i >> 5, d4 = (i & 31) << 2;
            v4f x = *(const v4f*)&X[((size_t)(tileB + r) * 200 + t) * 128 + d4];
            v4h xh = __builtin_convertvector(x, v4h);
            *(v4h*)&sXH[r][d4] = xh;
            *(v4h*)&sXR[r][d4] = xh;
        }
        if (t + 1 < 200) {
            const int pr = tid >> 4, pd = (tid & 15) * 8;
            __builtin_prefetch(&X[((size_t)(tileB + pr) * 200 + t + 1) * 128 + pd], 0, 1);
        }
#endif
        __syncthreads();

        // gates = sigmoid([x|h] @ Wg + bg)
        v8f accg[2] = {};
#pragma unroll
        for (int kb = 0; kb < 8; ++kb) {
            v16h a = load_A(&sXH[0][0], 256, kb * 32);
            accg[0] = WMMA_F16(a, bgf[kb][0], accg[0]);
            accg[1] = WMMA_F16(a, bgf[kb][1], accg[1]);
        }
#pragma unroll
        for (int j = 0; j < 2; ++j) {
            const int   ncol = wave * 32 + j * 16 + nloc;   // 0..255
            const float bias = j ? biasg1 : biasg0;
#pragma unroll
            for (int r = 0; r < 8; ++r) {
                const int   M = r + Mhi;
                const float g = sigmoidf_(accg[j][r] + bias);
                if (ncol < 128) {                       // reset gate -> rg*h
                    sXR[M][128 + ncol] = (_Float16)(g * sH[M][ncol]);
                } else {                                // update gate
                    sU[M][ncol - 128] = g;
                }
            }
        }
        __syncthreads();

        // c = tanh([x|rg*h] @ Wc + bc)
        v8f accc = {};
#pragma unroll
        for (int kb = 0; kb < 8; ++kb) {
            v16h a = load_A(&sXR[0][0], 256, kb * 32);
            accc = WMMA_F16(a, bcf[kb], accc);
        }
        {
            const int ncol = wave * 16 + nloc;
#pragma unroll
            for (int r = 0; r < 8; ++r)
                sC[r + Mhi][ncol] = tanhf(accc[r] + biasc);
        }
        __syncthreads();

        // h' = u*h + (1-u)*c, with seq_len copy-through and zeroed outputs
        for (int i = tid; i < 512; i += 256) {
            const int r = i >> 5, d4 = (i & 31) << 2;
            const int b = tileB + r;
            v4f u = *(const v4f*)&sU[r][d4];
            if (AUGRU) u = u * (1.f - att[(size_t)b * 200 + t]);
            v4f h = *(const v4f*)&sH[r][d4];
            v4f c = *(const v4f*)&sC[r][d4];
            v4f hn = u * h + (1.f - u) * c;
            const bool valid = t < seq_len[b];
            if (!valid) hn = h;
            *(v4f*)&sH[r][d4] = hn;
            *(v4h*)&sXH[r][128 + d4] = __builtin_convertvector(hn, v4h);
            if (out_seq) {
                v4f o = valid ? hn : (v4f){0.f, 0.f, 0.f, 0.f};
                *(v4f*)&out_seq[((size_t)b * 200 + t) * 128 + d4] = o;
            }
        }
        __syncthreads();
    }

    if (out_final) {
        for (int i = tid; i < 512; i += 256) {
            const int r = i >> 5, d4 = (i & 31) << 2;
            *(v4f*)&out_final[(size_t)(tileB + r) * 128 + d4] = *(const v4f*)&sH[r][d4];
        }
    }
}

// ---------------------------------------------------------------------------
// Kernel 2: DIN attention scorer.  One 16-row M-tile (rows of B*T) per wave,
// 2 waves per block.  din_all = [q, r, q-r, q*r] staged as [q|r] and [q-r|q*r].
// ---------------------------------------------------------------------------
__global__ __launch_bounds__(64)
void attn_kernel(const float* __restrict__ item_eb,  // [B,128]
                 const float* __restrict__ rnn1,     // [B,200,128]
                 const _Float16* __restrict__ pW1,   // packed [16kb][5nt]
                 const float* __restrict__ b1,
                 const _Float16* __restrict__ pW2,   // packed [3kb][3nt]
                 const float* __restrict__ b2,
                 const float* __restrict__ w3, const float* __restrict__ b3,
                 const int*   __restrict__ seq_len,
                 float* __restrict__ scores)         // [B,200]
{
    const int wave = threadIdx.x >> 5;
    const int lane = threadIdx.x & 31;
    const int tile = blockIdx.x * 2 + wave;   // 0..6399
    const int row0 = tile * 16;               // global row = b*200 + t
    const int Mhi  = (lane >> 4) << 3;
    const int nloc = lane & 15;

    __shared__ _Float16 sQR[2][16][256];   // [q | r]
    __shared__ _Float16 sDM[2][16][256];   // [q-r | q*r]
    __shared__ _Float16 sH1[2][16][96];
    __shared__ float    sH2[2][16][40];

    for (int i = lane; i < 512; i += 32) {
        const int r = i >> 5, d4 = (i & 31) << 2;
        const int R = row0 + r;
        const int b = R / 200, tt = R % 200;
        v4f q = *(const v4f*)&item_eb[(size_t)b * 128 + d4];
        v4f h = *(const v4f*)&rnn1[(size_t)(b * 200 + tt) * 128 + d4];
        *(v4h*)&sQR[wave][r][d4]       = __builtin_convertvector(q, v4h);
        *(v4h*)&sQR[wave][r][128 + d4] = __builtin_convertvector(h, v4h);
        *(v4h*)&sDM[wave][r][d4]       = __builtin_convertvector(q - h, v4h);
        *(v4h*)&sDM[wave][r][128 + d4] = __builtin_convertvector(q * h, v4h);
    }
    for (int i = lane; i < 16 * 96 / 4; i += 32)
        ((v4h*)sH1[wave])[i] = (v4h)(_Float16)0.f;
    __syncthreads();

    // layer 1: [16,512] @ [512,80] -> sigmoid
    v8f acc1[5] = {};
#pragma unroll
    for (int kb = 0; kb < 16; ++kb) {
        v16h a = (kb < 8) ? load_A(&sQR[wave][0][0], 256, kb * 32)
                          : load_A(&sDM[wave][0][0], 256, (kb - 8) * 32);
#pragma unroll
        for (int n = 0; n < 5; ++n)
            acc1[n] = WMMA_F16(a, load_B_packed(pW1, 5, kb, n), acc1[n]);
    }
#pragma unroll
    for (int n = 0; n < 5; ++n) {
        const int   ncol = n * 16 + nloc;      // < 80
        const float bias = b1[ncol];
#pragma unroll
        for (int r = 0; r < 8; ++r)
            sH1[wave][r + Mhi][ncol] = (_Float16)sigmoidf_(acc1[n][r] + bias);
    }
    __syncthreads();

    // layer 2: [16,80(pad96)] @ [80,40(pad48)] -> sigmoid
    v8f acc2[3] = {};
#pragma unroll
    for (int kb = 0; kb < 3; ++kb) {
        v16h a = load_A(&sH1[wave][0][0], 96, kb * 32);
#pragma unroll
        for (int n = 0; n < 3; ++n)
            acc2[n] = WMMA_F16(a, load_B_packed(pW2, 3, kb, n), acc2[n]);
    }
#pragma unroll
    for (int n = 0; n < 3; ++n) {
        const int ncol = n * 16 + nloc;
        if (ncol < 40) {
            const float bias = b2[ncol];
#pragma unroll
            for (int r = 0; r < 8; ++r)
                sH2[wave][r + Mhi][ncol] = sigmoidf_(acc2[n][r] + bias);
        }
    }
    __syncthreads();

    // layer 3: [16,40] @ [40,1] + mask
    if (lane < 16) {
        const int R = row0 + lane;
        const int b = R / 200, t = R % 200;
        float s = b3[0];
        for (int j = 0; j < 40; ++j) s += sH2[wave][lane][j] * w3[j];
        const bool valid = t < seq_len[b];
        scores[R] = valid ? s : -4294967295.0f;   // DIEN padding value -(2^32)+1
    }
}

// ---------------------------------------------------------------------------
// Kernel 3: masked softmax over T (scores already hold NEG_INF at pads)
// ---------------------------------------------------------------------------
__global__ __launch_bounds__(256)
void softmax_T(const float* __restrict__ scores, float* __restrict__ alphas) {
    __shared__ float red[256];
    const int b = blockIdx.x, tid = threadIdx.x;
    const float s = (tid < 200) ? scores[(size_t)b * 200 + tid] : -3.0e38f;
    red[tid] = s;
    __syncthreads();
    for (int k = 128; k > 0; k >>= 1) {
        if (tid < k) red[tid] = fmaxf(red[tid], red[tid + k]);
        __syncthreads();
    }
    const float m = red[0];
    __syncthreads();
    const float e = (tid < 200) ? __expf(s - m) : 0.f;
    red[tid] = e;
    __syncthreads();
    for (int k = 128; k > 0; k >>= 1) {
        if (tid < k) red[tid] += red[tid + k];
        __syncthreads();
    }
    if (tid < 200) alphas[(size_t)b * 200 + tid] = e / red[0];
}

// ---------------------------------------------------------------------------
// Kernel 5: masked history sum
// ---------------------------------------------------------------------------
__global__ __launch_bounds__(128)
void his_sum_kernel(const float* __restrict__ his, const int* __restrict__ seq_len,
                    float* __restrict__ hsum) {
    const int b = blockIdx.x, d = threadIdx.x;
    const int L = seq_len[b];
    float s = 0.f;
    for (int t = 0; t < L; ++t) s += his[((size_t)b * 200 + t) * 128 + d];
    hsum[(size_t)b * 128 + d] = s;
}

// ---------------------------------------------------------------------------
// Kernel 6: FC head (512 -> 200 -> 80 -> 2) + softmax.  One wave per 16 rows.
// ---------------------------------------------------------------------------
__global__ __launch_bounds__(32)
void fc_head(const float* __restrict__ item_eb, const float* __restrict__ hsum,
             const float* __restrict__ final2,
             const _Float16* __restrict__ pF1,   // packed [16kb][13nt]
             const float* __restrict__ f1b, const float* __restrict__ a1,
             const _Float16* __restrict__ pF2,   // packed [7kb][5nt]
             const float* __restrict__ f2b, const float* __restrict__ a2,
             const float* __restrict__ f3w, const float* __restrict__ f3b,
             float* __restrict__ out) {
    __shared__ _Float16 sF[16][512];
    __shared__ _Float16 sZ1[16][224];
    __shared__ float    sZ2[16][80];
    const int lane = threadIdx.x;
    const int b0   = blockIdx.x * 16;
    const int Mhi  = (lane >> 4) << 3;
    const int nloc = lane & 15;

    for (int i = lane; i < 2048; i += 32) {   // 16 rows x 512 cols, 4-wide
        const int r = i >> 7, c4 = (i & 127) << 2;
        const int b = b0 + r, d4 = c4 & 127;
        v4f v;
        switch (c4 >> 7) {
            case 0:  v = *(const v4f*)&item_eb[(size_t)b * 128 + d4]; break;
            case 1:  v = *(const v4f*)&hsum[(size_t)b * 128 + d4]; break;
            case 2:  v = *(const v4f*)&item_eb[(size_t)b * 128 + d4] *
                         *(const v4f*)&hsum[(size_t)b * 128 + d4]; break;
            default: v = *(const v4f*)&final2[(size_t)b * 128 + d4]; break;
        }
        *(v4h*)&sF[r][c4] = __builtin_convertvector(v, v4h);
    }
    for (int i = lane; i < 16 * 224 / 4; i += 32) ((v4h*)sZ1)[i] = (v4h)(_Float16)0.f;
    __syncthreads();

    // fc1: [16,512] @ [512,200(pad208)] -> PReLU
    v8f accz[13] = {};
#pragma unroll
    for (int kb = 0; kb < 16; ++kb) {
        v16h a = load_A(&sF[0][0], 512, kb * 32);
#pragma unroll
        for (int n = 0; n < 13; ++n)
            accz[n] = WMMA_F16(a, load_B_packed(pF1, 13, kb, n), accz[n]);
    }
#pragma unroll
    for (int n = 0; n < 13; ++n) {
        const int ncol = n * 16 + nloc;
        if (ncol < 200) {
            const float bias = f1b[ncol], al = a1[ncol];
#pragma unroll
            for (int r = 0; r < 8; ++r) {
                const float x = accz[n][r] + bias;
                sZ1[r + Mhi][ncol] = (_Float16)(fmaxf(x, 0.f) + al * fminf(x, 0.f));
            }
        }
    }
    __syncthreads();

    // fc2: [16,200(pad224)] @ [200,80] -> PReLU
    v8f acc2[5] = {};
#pragma unroll
    for (int kb = 0; kb < 7; ++kb) {
        v16h a = load_A(&sZ1[0][0], 224, kb * 32);
#pragma unroll
        for (int n = 0; n < 5; ++n)
            acc2[n] = WMMA_F16(a, load_B_packed(pF2, 5, kb, n), acc2[n]);
    }
#pragma unroll
    for (int n = 0; n < 5; ++n) {
        const int   ncol = n * 16 + nloc;
        const float bias = f2b[ncol], al = a2[ncol];
#pragma unroll
        for (int r = 0; r < 8; ++r) {
            const float x = acc2[n][r] + bias;
            sZ2[r + Mhi][ncol] = fmaxf(x, 0.f) + al * fminf(x, 0.f);
        }
    }
    __syncthreads();

    // fc3 + 2-way softmax
    if (lane < 16) {
        float l0 = f3b[0], l1 = f3b[1];
        for (int j = 0; j < 80; ++j) {
            const float z = sZ2[lane][j];
            l0 += z * f3w[j * 2 + 0];
            l1 += z * f3w[j * 2 + 1];
        }
        const float m  = fmaxf(l0, l1);
        const float e0 = __expf(l0 - m), e1 = __expf(l1 - m);
        const float inv = 1.f / (e0 + e1);
        const int b = b0 + lane;
        out[(size_t)b * 2 + 0] = e0 * inv;
        out[(size_t)b * 2 + 1] = e1 * inv;
    }
}

// ---------------------------------------------------------------------------
// Host-side launcher
// ---------------------------------------------------------------------------
extern "C" void kernel_launch(void* const* d_in, const int* in_sizes, int n_in,
                              void* d_out, int out_size, void* d_ws, size_t ws_size,
                              hipStream_t stream) {
    const float* item_eb = (const float*)d_in[0];
    const float* his     = (const float*)d_in[1];
    const float* Wg1     = (const float*)d_in[2];
    const float* bg1     = (const float*)d_in[3];
    const float* Wc1     = (const float*)d_in[4];
    const float* bc1     = (const float*)d_in[5];
    const float* aw1     = (const float*)d_in[6];
    const float* ab1     = (const float*)d_in[7];
    const float* aw2     = (const float*)d_in[8];
    const float* ab2     = (const float*)d_in[9];
    const float* aw3     = (const float*)d_in[10];
    const float* ab3     = (const float*)d_in[11];
    const float* Wg2     = (const float*)d_in[12];
    const float* bg2     = (const float*)d_in[13];
    const float* Wc2     = (const float*)d_in[14];
    const float* bc2     = (const float*)d_in[15];
    const float* f1w     = (const float*)d_in[16];
    const float* f1b     = (const float*)d_in[17];
    const float* al1     = (const float*)d_in[18];
    const float* f2w     = (const float*)d_in[19];
    const float* f2b     = (const float*)d_in[20];
    const float* al2     = (const float*)d_in[21];
    const float* f3w     = (const float*)d_in[22];
    const float* f3b     = (const float*)d_in[23];
    const int*   seq     = (const int*)d_in[24];
    float*       out     = (float*)d_out;

    // workspace layout: f32 buffers, then packed f16 weight fragments
    float* rnn1   = (float*)d_ws;                              // 512*200*128
    float* scores = rnn1 + (size_t)512 * 200 * 128;            // 512*200
    float* alphas = scores + (size_t)512 * 200;                // 512*200
    float* final2 = alphas + (size_t)512 * 200;                // 512*128
    float* hsum   = final2 + (size_t)512 * 128;                // 512*128
    _Float16* packed = (_Float16*)(hsum + (size_t)512 * 128);
    _Float16* pWg1 = packed;                 // 8kb*16nt*512 = 65536
    _Float16* pWc1 = pWg1 + 65536;           // 8kb* 8nt*512 = 32768
    _Float16* pW1  = pWc1 + 32768;           // 16kb*5nt*512 = 40960
    _Float16* pW2  = pW1 + 40960;            // 3kb*3nt*512  = 4608
    _Float16* pWg2 = pW2 + 4608;             // 65536
    _Float16* pWc2 = pWg2 + 65536;           // 32768
    _Float16* pF1  = pWc2 + 32768;           // 16kb*13nt*512 = 106496
    _Float16* pF2  = pF1 + 106496;           // 7kb*5nt*512   = 17920

    // 0) pre-swizzle weights into WMMA B-fragment layout (f16)
    pack_B<<<128, 256, 0, stream>>>(Wg1, 256, 256, 8, 16, pWg1);
    pack_B<<<64, 256, 0, stream>>>(Wc1, 256, 128, 8, 8, pWc1);
    pack_B<<<80, 256, 0, stream>>>(aw1, 512, 80, 16, 5, pW1);
    pack_B<<<9, 256, 0, stream>>>(aw2, 80, 40, 3, 3, pW2);
    pack_B<<<128, 256, 0, stream>>>(Wg2, 256, 256, 8, 16, pWg2);
    pack_B<<<64, 256, 0, stream>>>(Wc2, 256, 128, 8, 8, pWc2);
    pack_B<<<208, 256, 0, stream>>>(f1w, 512, 200, 16, 13, pF1);
    pack_B<<<35, 256, 0, stream>>>(f2w, 200, 80, 7, 5, pF2);

    // 1) GRU over history
    gru_kernel<false><<<32, 256, 0, stream>>>(his, pWg1, bg1, pWc1, bc1, seq,
                                              nullptr, rnn1, nullptr);
    // 2) DIN attention scores
    attn_kernel<<<3200, 64, 0, stream>>>(item_eb, rnn1, pW1, ab1, pW2, ab2,
                                         aw3, ab3, seq, scores);
    // 3) masked softmax over T
    softmax_T<<<512, 256, 0, stream>>>(scores, alphas);
    // 4) AUGRU -> final state
    gru_kernel<true><<<32, 256, 0, stream>>>(rnn1, pWg2, bg2, pWc2, bc2, seq,
                                             alphas, nullptr, final2);
    // 5) masked history sum
    his_sum_kernel<<<512, 128, 0, stream>>>(his, seq, hsum);
    // 6) FC head + output softmax
    fc_head<<<32, 32, 0, stream>>>(item_eb, hsum, final2, pF1, f1b, al1,
                                   pF2, f2b, al2, f3w, f3b, out);
}
